// CERNNModel_8048768713387
// MI455X (gfx1250) — matrix operand order repaired and checked by reading, not compiled, not run on hardware
//
#include <hip/hip_runtime.h>
#include <hip/hip_bf16.h>
#include <cstddef>

typedef unsigned short ushort_t;
typedef __attribute__((ext_vector_type(16))) __bf16 v16bf;
typedef __attribute__((ext_vector_type(8)))  float  v8f;

#define N_RNN   4096
#define N_INPUT 85
#define N_OUT   33
#define T_LEN   256
#define B_SZ    32
#define HID_ELEMS (B_SZ * N_RNN)          // 131072 per timestep
#define N_TILES_N (N_RNN / 16)            // 256
#define N_KBLK    (N_RNN / 32)            // 128
#define TILES_TOTAL (2 * N_TILES_N)       // 512 output tiles (2 M x 256 N)
#define SPLITK  8                         // waves per tile
#define KB_PER_WAVE (N_KBLK / SPLITK)     // 16

__device__ __forceinline__ ushort_t f2bf(float f) {
    unsigned u = __float_as_uint(f);
    u += 0x7fffu + ((u >> 16) & 1u);      // round-to-nearest-even
    return (ushort_t)(u >> 16);
}

__device__ __forceinline__ float input_mask(int h, int i) {
    bool m = (i >= 5) && (i <= 83);
    if (h < 64)            m = m || (i == 0) || (i == 3) || (i == 4);
    else if (h < 128)      m = m || (i == 1) || (i == 2);
    return m ? 1.0f : 0.0f;
}

// ---------------- preprocessing ----------------

__global__ void init_kernel(float* h0, ushort_t* fragH0, int* flags) {
    int i = blockIdx.x * blockDim.x + threadIdx.x;
    if (i < HID_ELEMS) { h0[i] = 0.0f; fragH0[i] = 0; }
    if (i < 4) flags[i] = 0;
}

__global__ void wih_kernel(const float* __restrict__ w_ih, float* __restrict__ wih) {
    int idx = blockIdx.x * blockDim.x + threadIdx.x;
    if (idx >= N_RNN * N_INPUT) return;
    int h = idx / N_INPUT, i = idx - h * N_INPUT;
    wih[idx] = w_ih[idx] * input_mask(h, i);
}

// Pre-swizzle Wm (masked+thresholded weight_hh) into B-fragment layout:
// Bfrag[((ntile*128 + kblock)*32 + lane)*16 + e] = bf16(Wm[n][k])
// with n = ntile*16 + (lane&15), k = kblock*32 + (lane>>4)*16 + e.
__global__ void bfrag_kernel(const float* __restrict__ whh,
                             const float* __restrict__ imask,
                             ushort_t* __restrict__ Bfrag) {
    int tid = blockIdx.x * blockDim.x + threadIdx.x;      // = (ntile*128+kb)*32+lane
    if (tid >= N_TILES_N * N_KBLK * 32) return;
    int lane  = tid & 31;
    int kb    = (tid >> 5) & (N_KBLK - 1);
    int ntile = tid >> 12;
    int n     = ntile * 16 + (lane & 15);
    int kbase = kb * 32 + (lane >> 4) * 16;
    ushort_t* dst = Bfrag + (size_t)tid * 16;
    const float* wrow = whh + (size_t)n * N_RNN;
    const float* mrow = imask + (size_t)n * N_RNN;
    #pragma unroll
    for (int e = 0; e < 16; ++e) {
        int k = kbase + e;
        float w = wrow[k];
        float keep = (fabsf(w) > 0.01f) ? mrow[k] : 0.0f;
        dst[e] = f2bf(w * keep);
    }
}

// i0/i1 broadcast vectors + per-neuron dopamine multiplier
__global__ void ivec_kernel(const float* __restrict__ x, const float* __restrict__ wih,
                            const float* __restrict__ d1,
                            float* __restrict__ i0, float* __restrict__ i1,
                            float* __restrict__ mult) {
    int h = blockIdx.x * blockDim.x + threadIdx.x;
    if (h >= N_RNN) return;
    const float* w  = wih + (size_t)h * N_INPUT;
    const float* x0 = x;                          // x[0, 0, :]
    const float* x1 = x + (size_t)B_SZ * N_INPUT; // x[1, 0, :]
    float s0 = 0.0f, s1 = 0.0f;
    for (int i = 0; i < N_INPUT; ++i) { s0 += x0[i] * w[i]; s1 += x1[i] * w[i]; }
    i0[h] = s0; i1[h] = s1;
    mult[h] = 1.0f + 0.4f * (d1[h] - 0.5f);
}

// inp_seq[t,b,:] = x[t,b,:] @ wih^T  -> written into hidden slots of d_out
__global__ void inp_kernel(const float* __restrict__ x, const float* __restrict__ wih,
                           float* __restrict__ hidden) {
    __shared__ float xr[N_INPUT];
    int tb = blockIdx.x;                      // t*B + b
    if (threadIdx.x < N_INPUT) xr[threadIdx.x] = x[(size_t)tb * N_INPUT + threadIdx.x];
    __syncthreads();
    for (int h = threadIdx.x; h < N_RNN; h += blockDim.x) {
        const float* w = wih + (size_t)h * N_INPUT;
        float s = 0.0f;
        #pragma unroll 5
        for (int i = 0; i < N_INPUT; ++i) s += xr[i] * w[i];
        hidden[(size_t)tb * N_RNN + h] = s;
    }
}

// ---------------- recurrent cell (the WMMA workhorse) ----------------
// Split-K: one 256-thread block (8 waves) per 16x16 output tile. Wave w
// accumulates 16 K-blocks with v_wmma_f32_16x16x32_bf16, partials are
// reduced through LDS, then 256 threads apply the leaky-integrate + relu +
// dopamine epilogue (one element each), writing h_new (fp32) and scattering
// it into next step's A-fragment (bf16) layout.
__global__ __launch_bounds__(256) void cell_kernel(
        const ushort_t* __restrict__ Bfrag,
        const ushort_t* __restrict__ fragIn,
        ushort_t* __restrict__ fragOut,
        const float* hold,
        float* hout,
        const float* __restrict__ inp_vec,    // broadcast mode (fixed point)
        const float* inp_mat,                 // per-element mode (scan); may alias hout
        const float* __restrict__ bias,
        const float* __restrict__ mult,
        const int* done_flag, int use_done) {
    if (use_done && done_flag[2]) return;     // uniform -> EXEC stays all-ones
    __shared__ float red[SPLITK][32][9];      // 9-float stride: bank-conflict free

    int lane = threadIdx.x & 31;
    int wv   = threadIdx.x >> 5;              // 0..7 split-K slice
    int tile = blockIdx.x;                    // 0..511
    int mtile = tile & 1;                     // M = 32 -> 2 tiles
    int ntile = tile >> 1;                    // 0..255

    const v16bf* Bp = (const v16bf*)Bfrag +
                      ((size_t)ntile * N_KBLK + (size_t)wv * KB_PER_WAVE) * 32 + lane;
    const v16bf* Ap = (const v16bf*)fragIn +
                      ((size_t)mtile * N_KBLK + (size_t)wv * KB_PER_WAVE) * 32 + lane;

    v8f acc = {};
    #pragma unroll 4
    for (int kb = 0; kb < KB_PER_WAVE; ++kb) {
        v16bf a = Ap[(size_t)kb * 32];
        v16bf b = Bp[(size_t)kb * 32];
        acc = __builtin_amdgcn_wmma_f32_16x16x32_bf16(
                  false, a, false, b, (short)0, acc, false, false);
    }
    #pragma unroll
    for (int r = 0; r < 8; ++r) red[wv][lane][r] = acc[r];
    __syncthreads();

    // epilogue: thread i owns element (M = i>>4, N = i&15) of the tile
    int i = threadIdx.x;
    int M = i >> 4, N = i & 15;
    int rlane = (M & 8) ? (16 + N) : N;       // C/D layout: lanes16-31 hold M=8..15
    int rr    = M & 7;
    float s = 0.0f;
    #pragma unroll
    for (int w = 0; w < SPLITK; ++w) s += red[w][rlane][rr];

    int n = ntile * 16 + N;
    int m = mtile * 16 + M;
    size_t idx = (size_t)m * N_RNN + n;
    float inp = inp_vec ? inp_vec[n] : inp_mat[idx];
    float ho  = hold[idx];
    float pre = s + inp + bias[n];
    float rl  = fmaxf(pre, 0.0f);
    float hn  = (0.8f * ho + 0.2f * rl) * mult[n];
    hout[idx] = hn;

    // scatter into next step's A-fragment layout
    int kb2   = n >> 5;
    int kin   = n & 31;
    int khalf = kin >> 4;
    int e     = kin & 15;
    int ldst  = M + 16 * khalf;               // m & 15 == M
    fragOut[(((size_t)mtile * N_KBLK + kb2) * 32 + ldst) * 16 + e] = f2bf(hn);
}

// ---------------- fixed-point control ----------------

__global__ void viol_kernel(const float* __restrict__ hnew,
                            const float* __restrict__ h0, int* flags) {
    if (flags[2]) return;
    int i = blockIdx.x * blockDim.x + threadIdx.x;
    if (i >= HID_ELEMS) return;
    float a = hnew[i], b = h0[i];
    if (fabsf(a - b) > 0.1f + 1e-5f * fabsf(b)) atomicOr(&flags[0], 1);
}

__global__ void update_kernel(int* flags) {
    if (!flags[2]) {
        int close = (flags[0] == 0);
        flags[1] = close ? flags[1] + 1 : 0;
        if (flags[1] >= 4) flags[2] = 1;
    }
    flags[0] = 0;
}

__global__ void copy_kernel(float* __restrict__ h0, const float* __restrict__ hnew,
                            ushort_t* __restrict__ fragH0,
                            const ushort_t* __restrict__ fragNew, const int* flags) {
    if (flags[2]) return;
    int i = blockIdx.x * blockDim.x + threadIdx.x;
    if (i >= HID_ELEMS) return;
    h0[i] = hnew[i];
    fragH0[i] = fragNew[i];
}

// ---------------- output projection ----------------

__global__ void out_kernel(const float* __restrict__ hidden,
                           const float* __restrict__ w_out,
                           float* __restrict__ out) {
    int idx = blockIdx.x * blockDim.x + threadIdx.x;
    if (idx >= T_LEN * B_SZ * N_OUT) return;
    int o = idx % N_OUT;
    int tb = idx / N_OUT;
    const float* hr = hidden + (size_t)tb * N_RNN + 128;   // MOTOR_START
    const float* wr = w_out + (size_t)o * 64;
    float s = 0.0f;
    #pragma unroll 8
    for (int j = 0; j < 64; ++j) s += hr[j] * wr[j];
    out[idx] = s;
}

// ---------------- host driver ----------------

extern "C" void kernel_launch(void* const* d_in, const int* in_sizes, int n_in,
                              void* d_out, int out_size, void* d_ws, size_t ws_size,
                              hipStream_t stream) {
    const float* x      = (const float*)d_in[0];
    const float* w_ih   = (const float*)d_in[1];
    const float* w_hh   = (const float*)d_in[2];
    const float* bias   = (const float*)d_in[3];
    const float* w_out  = (const float*)d_in[4];
    const float* d1     = (const float*)d_in[5];
    const float* imask  = (const float*)d_in[6];
    float* out = (float*)d_out;

    // workspace layout (all offsets 64B aligned)
    char* ws = (char*)d_ws;
    const size_t B_OFF    = 0;                         // 4096*4096 bf16 = 32 MiB
    const size_t FH0_OFF  = B_OFF   + (size_t)N_RNN * N_RNN * 2;
    const size_t FT1_OFF  = FH0_OFF + (size_t)HID_ELEMS * 2;
    const size_t FT2_OFF  = FT1_OFF + (size_t)HID_ELEMS * 2;
    const size_t H0_OFF   = FT2_OFF + (size_t)HID_ELEMS * 2;
    const size_t HT_OFF   = H0_OFF  + (size_t)HID_ELEMS * 4;
    const size_t HN_OFF   = HT_OFF  + (size_t)HID_ELEMS * 4;
    const size_t WIH_OFF  = HN_OFF  + (size_t)HID_ELEMS * 4;
    const size_t I0_OFF   = WIH_OFF + (size_t)N_RNN * N_INPUT * 4 + 192; // extra 192B pad; offsets from here are 64B aligned
    const size_t I1_OFF   = I0_OFF  + (size_t)N_RNN * 4;
    const size_t MUL_OFF  = I1_OFF  + (size_t)N_RNN * 4;
    const size_t FLG_OFF  = MUL_OFF + (size_t)N_RNN * 4;
    const size_t NEED     = FLG_OFF + 256;
    if (ws_size < NEED) return;

    ushort_t* Bfrag  = (ushort_t*)(ws + B_OFF);
    ushort_t* fragH0 = (ushort_t*)(ws + FH0_OFF);
    ushort_t* fragT1 = (ushort_t*)(ws + FT1_OFF);
    ushort_t* fragT2 = (ushort_t*)(ws + FT2_OFF);
    float* h0   = (float*)(ws + H0_OFF);
    float* htmp = (float*)(ws + HT_OFF);
    float* hnew = (float*)(ws + HN_OFF);
    float* wih  = (float*)(ws + WIH_OFF);
    float* i0v  = (float*)(ws + I0_OFF);
    float* i1v  = (float*)(ws + I1_OFF);
    float* mult = (float*)(ws + MUL_OFF);
    int*   flags = (int*)(ws + FLG_OFF);

    float* hidden = out + (size_t)T_LEN * B_SZ * N_OUT;   // hidden part of d_out

    // preprocessing
    init_kernel<<<(HID_ELEMS + 255) / 256, 256, 0, stream>>>(h0, fragH0, flags);
    wih_kernel<<<(N_RNN * N_INPUT + 255) / 256, 256, 0, stream>>>(w_ih, wih);
    bfrag_kernel<<<(N_TILES_N * N_KBLK * 32) / 256, 256, 0, stream>>>(w_hh, imask, Bfrag);
    ivec_kernel<<<N_RNN / 256, 256, 0, stream>>>(x, wih, d1, i0v, i1v, mult);
    inp_kernel<<<T_LEN * B_SZ, 256, 0, stream>>>(x, wih, hidden);

    const int CB = 256;                 // 8 waves/block: one tile, split-K=8
    const int CG = TILES_TOTAL;         // 512 blocks

    // fixed-point: 100 iterations of cell(cell(h0,i0),i1), gated on device flag
    for (int it = 0; it < 100; ++it) {
        cell_kernel<<<CG, CB, 0, stream>>>(Bfrag, fragH0, fragT1, h0, htmp,
                                           i0v, nullptr, bias, mult, flags, 1);
        cell_kernel<<<CG, CB, 0, stream>>>(Bfrag, fragT1, fragT2, htmp, hnew,
                                           i1v, nullptr, bias, mult, flags, 1);
        viol_kernel<<<(HID_ELEMS + 255) / 256, 256, 0, stream>>>(hnew, h0, flags);
        update_kernel<<<1, 1, 0, stream>>>(flags);
        copy_kernel<<<(HID_ELEMS + 255) / 256, 256, 0, stream>>>(h0, hnew, fragH0, fragT2, flags);
    }

    // scan over T timesteps; inp[t] lives in hidden slot t and is overwritten by h_t
    for (int t = 0; t < T_LEN; ++t) {
        const ushort_t* fin = (t == 0) ? fragH0 : ((t & 1) ? fragT1 : fragT2);
        ushort_t* fot = (t & 1) ? fragT2 : fragT1;
        const float* hold = (t == 0) ? h0 : hidden + (size_t)(t - 1) * HID_ELEMS;
        float* slot = hidden + (size_t)t * HID_ELEMS;
        cell_kernel<<<CG, CB, 0, stream>>>(Bfrag, fin, fot, hold, slot,
                                           nullptr, slot, bias, mult, nullptr, 0);
    }

    out_kernel<<<(T_LEN * B_SZ * N_OUT + 255) / 256, 256, 0, stream>>>(hidden, w_out, out);
}
